// NeuralGraphHidden_39049842655949
// MI455X (gfx1250) — matrix operand order repaired and checked by reading, not compiled, not run on hardware
//
#include <hip/hip_runtime.h>
#include <hip/hip_bf16.h>

typedef _Float16 v16h __attribute__((ext_vector_type(16)));
typedef _Float16 v8h  __attribute__((ext_vector_type(8)));
typedef float    v8f  __attribute__((ext_vector_type(8)));

#define NBATCH 256
#define NATOM  128
#define ND     5
#define FATOM  256
#define FBOND  64
#define FK     320          // FATOM + FBOND
#define NC     256
#define NROWS  (NBATCH*NATOM)   // 32768
#define CAP    (NROWS + 16)     // per-degree list capacity (room for pad)
#define NTILES (NROWS/16)       // 2048
#define LDSS   328              // LDS feat row stride in halves (conflict-free b128)

// ---------------------------------------------------------------------------
// Kernel 1: convert W (d,f,c) f32 -> Wt (d,c,f) f16   (5*256*320 elements)
// ---------------------------------------------------------------------------
__global__ __launch_bounds__(256) void ngh_wprep(const float* __restrict__ W,
                                                 _Float16* __restrict__ Wt) {
    int idx = blockIdx.x * 256 + threadIdx.x;           // over d*c*f, f fastest
    if (idx >= ND * NC * FK) return;
    int f = idx % FK;
    int c = (idx / FK) % NC;
    int d = idx / (FK * NC);
    Wt[idx] = (_Float16)W[((size_t)d * FK + f) * NC + c];
}

// ---------------------------------------------------------------------------
// Kernel 2: bin rows by degree (deg = #edges != -1); deg==5 rows are dropped
// (their output is all-zero and handled by the memset).
// ---------------------------------------------------------------------------
__global__ __launch_bounds__(256) void ngh_groups(const int* __restrict__ edges,
                                                  int* __restrict__ cnt,
                                                  int* __restrict__ lists) {
    int row = blockIdx.x * 256 + threadIdx.x;
    if (row >= NROWS) return;
    int deg = 0;
#pragma unroll
    for (int d = 0; d < ND; ++d) deg += (edges[row * ND + d] != -1) ? 1 : 0;
    if (deg < ND) {
        int pos = atomicAdd(&cnt[deg], 1);
        lists[deg * CAP + pos] = row;
    }
}

// ---------------------------------------------------------------------------
// Kernel 3: pad each degree list to a multiple of 16 with sentinel -1 rows
// ---------------------------------------------------------------------------
__global__ void ngh_pad(const int* __restrict__ cnt, int* __restrict__ pcnt,
                        int* __restrict__ lists) {
    int d = threadIdx.x;
    if (d < ND) {
        int c = cnt[d];
        int p = (c + 15) & ~15;
        for (int i = c; i < p; ++i) lists[d * CAP + i] = -1;
        pcnt[d] = p;
    }
}

// ---------------------------------------------------------------------------
// Kernel 4: grouped GEMM. grid = ND * NTILES blocks; block g handles degree
// g/NTILES, row-tile g%NTILES of that degree's (padded) list. 128 threads =
// 4 waves; wave w computes output columns [w*64, w*64+64) for the 16 rows.
// Phase A: gather feat (self + neighbors + bond sums) into LDS as f16.
// Phase B: 10 x v_wmma_f32_16x16x32_f16 per 16-col subtile (40 WMMA / wave).
// ---------------------------------------------------------------------------
__global__ __launch_bounds__(128) void ngh_gemm(const float* __restrict__ atoms,
                                                const float* __restrict__ bonds,
                                                const int*   __restrict__ edges,
                                                const float* __restrict__ bias,
                                                const _Float16* __restrict__ Wt,
                                                const int* __restrict__ lists,
                                                const int* __restrict__ pcnt,
                                                float* __restrict__ out) {
    __shared__ __align__(16) _Float16 sfeat[16 * LDSS];
    __shared__ int srow[16];
    __shared__ int sptr[16][6];   // element base into atoms: self + 5 neighbors (-1 = none)

    int blk  = blockIdx.x;
    int d    = blk / NTILES;
    int tile = blk % NTILES;
    if (tile * 16 >= pcnt[d]) return;     // uniform per block -> full-EXEC waves below

    int t = threadIdx.x;
    if (t < 16) srow[t] = lists[d * CAP + tile * 16 + t];
    __syncthreads();

    if (t < 96) {
        int r = t / 6, j = t % 6;
        int row = srow[r];
        int p = -1;
        if (row >= 0) {
            if (j == 0) {
                p = row * FATOM;
            } else {
                int e = edges[row * ND + (j - 1)];
                if (e >= 0) p = ((row >> 7) * NATOM + e) * FATOM;   // NATOM = 128
            }
        }
        sptr[r][j] = p;
    }
    __syncthreads();

    // feat[0:256): atoms self + neighbor sum
    for (int i = t; i < 16 * FATOM; i += 128) {
        int r = i >> 8, f = i & (FATOM - 1);
        float v = 0.f;
        if (srow[r] >= 0) {
#pragma unroll
            for (int j = 0; j < 6; ++j) {
                int p = sptr[r][j];
                if (p >= 0) v += atoms[p + f];
            }
        }
        sfeat[r * LDSS + f] = (_Float16)v;
    }
    // feat[256:320): bond sum over D
    for (int i = t; i < 16 * FBOND; i += 128) {
        int r = i >> 6, f = i & (FBOND - 1);
        float v = 0.f;
        int row = srow[r];
        if (row >= 0) {
#pragma unroll
            for (int dd = 0; dd < ND; ++dd)
                v += bonds[(size_t)(row * ND + dd) * FBOND + f];
        }
        sfeat[r * LDSS + FATOM + f] = (_Float16)v;
    }
    __syncthreads();

    int wave  = t >> 5;
    int lane  = t & 31;
    int lhi   = (lane >= 16) ? 1 : 0;     // lane half select
    int ln    = lane & 15;
    int nbase = wave * 64;

    v8f acc[4] = {};                      // 4 x 16x16 f32 accumulators
    const _Float16* wb = Wt + (size_t)d * NC * FK;

    // A fragment addressing (16x32 f16): lanes 0-15 hold K{0..15},
    // lanes 16-31 hold K{16..31} of row = lane%16.
    const _Float16* abase = &sfeat[ln * LDSS + lhi * 16];

#pragma unroll 2
    for (int k0 = 0; k0 < FK; k0 += 32) {
        v8h lo = *(const v8h*)(abase + k0);
        v8h hi = *(const v8h*)(abase + k0 + 8);
        v16h a = __builtin_shufflevector(lo, hi, 0, 1, 2, 3, 4, 5, 6, 7,
                                                  8, 9, 10, 11, 12, 13, 14, 15);
#pragma unroll
        for (int st = 0; st < 4; ++st) {
            // B fragment (32x16 f16): lane holds column N = nbase+st*16+ln,
            // K range [k0 + lhi*16, +16) -> 32 contiguous bytes of Wt[d][c][*].
            const _Float16* bp =
                wb + (size_t)(nbase + st * 16 + ln) * FK + k0 + lhi * 16;
            v16h b = *(const v16h*)bp;
            acc[st] = __builtin_amdgcn_wmma_f32_16x16x32_f16(
                false, a, false, b, (short)0, acc[st], false, false);
        }
    }

    // epilogue: bias + relu, scatter rows (skip -1 padding rows)
#pragma unroll
    for (int st = 0; st < 4; ++st) {
        int col = nbase + st * 16 + ln;
        float bi = bias[d * NC + col];
#pragma unroll
        for (int r = 0; r < 8; ++r) {
            int row = srow[r + lhi * 8];   // VGPR r holds M=r (lo) / M=r+8 (hi)
            if (row >= 0) {
                float v = acc[st][r] + bi;
                out[(size_t)row * NC + col] = v > 0.f ? v : 0.f;
            }
        }
    }
}

// ---------------------------------------------------------------------------
extern "C" void kernel_launch(void* const* d_in, const int* in_sizes, int n_in,
                              void* d_out, int out_size, void* d_ws, size_t ws_size,
                              hipStream_t stream) {
    (void)in_sizes; (void)n_in; (void)ws_size;
    const float* atoms = (const float*)d_in[0];
    const float* bonds = (const float*)d_in[1];
    const int*   edges = (const int*)d_in[2];
    const float* W     = (const float*)d_in[3];
    const float* bias  = (const float*)d_in[4];
    float* out = (float*)d_out;

    char* ws   = (char*)d_ws;
    int* cnt   = (int*)ws;                 // 5 counters (zeroed each call)
    int* pcnt  = cnt + 8;                  // 5 padded counts
    int* lists = (int*)(ws + 1024);        // 5 * CAP row indices
    size_t wt_off = (1024 + (size_t)ND * CAP * 4 + 255) & ~(size_t)255;
    _Float16* Wt = (_Float16*)(ws + wt_off);   // 5*256*320 f16 (c-major)

    hipMemsetAsync(cnt, 0, 64, stream);
    hipMemsetAsync(out, 0, (size_t)out_size * sizeof(float), stream);

    ngh_wprep <<<(ND * NC * FK + 255) / 256, 256, 0, stream>>>(W, Wt);
    ngh_groups<<<NROWS / 256, 256, 0, stream>>>(edges, cnt, lists);
    ngh_pad   <<<1, 32, 0, stream>>>(cnt, pcnt, lists);
    ngh_gemm  <<<ND * NTILES, 128, 0, stream>>>(atoms, bonds, edges, bias,
                                                Wt, lists, pcnt, out);
}
